// MultiHeadAttention_85109071937989
// MI455X (gfx1250) — matrix operand order repaired and checked by reading, not compiled, not run on hardware
//
#include <hip/hip_runtime.h>
#include <hip/hip_bf16.h>

#define EMB 384
#define NH  6
#define HD  64
#define BBATCH 256
#define TSEQ 256

typedef __attribute__((ext_vector_type(16))) __bf16 v16bf;
typedef __attribute__((ext_vector_type(8)))  __bf16 v8bf;
typedef __attribute__((ext_vector_type(4)))  __bf16 v4bf;
typedef __attribute__((ext_vector_type(8)))  float  v8f;

// ---------------------------------------------------------------------------
// WMMA fragment helpers (CDNA5 wave32 layouts, CDNA5 ISA §7.12.2)
// A 16x32 bf16 (this code's contiguous K order): lane<16 -> row=lane, K=0..15; lane>=16 -> row=lane-16,
//               K=16..31
// B 32x16 bf16 (same K order as A): lane<16 -> col=lane, Krows 0..15; lane>=16 -> col=lane-16,
//               Krows 16..31
// C/D 16x16 f32: N=lane&15, M=vgpr + 8*(lane>=16)
// All loaders below are pure 16-byte vector loads (b128 / ds_load_b128).
// ---------------------------------------------------------------------------

__device__ inline v8f wmma_bf16(v16bf a, v16bf b, v8f c) {
  return __builtin_amdgcn_wmma_f32_16x16x32_bf16(
      /*neg_a=*/false, a, /*neg_b=*/false, b,
      /*c_mod=*/(short)0, c, /*reuse_a=*/false, /*reuse_b=*/false);
}

// A fragment from row-major bf16 matrix (row stride ld, in elements).
// Requires p 16B-aligned and ld a multiple of 8.
__device__ inline v16bf load_a_bf16(const __bf16* p, int ld, int lane) {
  int r = lane & 15;
  int half = lane >> 4;
  const __bf16* row = p + (size_t)r * ld + half * 16;
  v8bf lo = *(const v8bf*)(row);       // K 16*half + 0..7
  v8bf hi = *(const v8bf*)(row + 8);   // K 16*half + 8..15
  return __builtin_shufflevector(lo, hi, 0, 1, 2, 3, 4, 5, 6, 7,
                                 8, 9, 10, 11, 12, 13, 14, 15);
}

// B fragment where the 16 K-values per lane are contiguous:
// element(krow, n) = p[n*ld + 16*half + j]
__device__ inline v16bf load_b_contig_bf16(const __bf16* p, int ld, int lane) {
  int n = lane & 15;
  int half = lane >> 4;
  const __bf16* col = p + (size_t)n * ld + half * 16;
  v8bf lo = *(const v8bf*)(col);
  v8bf hi = *(const v8bf*)(col + 8);
  return __builtin_shufflevector(lo, hi, 0, 1, 2, 3, 4, 5, 6, 7,
                                 8, 9, 10, 11, 12, 13, 14, 15);
}

// ---------------------------------------------------------------------------
// Pre-pass 0a: x fp32 -> bf16 (streaming, 4 elems/thread)
// ---------------------------------------------------------------------------
__global__ __launch_bounds__(256) void cvt_x_kernel(const float* __restrict__ x,
                                                    __bf16* __restrict__ xb) {
  int i = blockIdx.x * 256 + threadIdx.x;   // over (B*T*EMB)/4
  float4 f = ((const float4*)x)[i];
  v4bf o;
  o[0] = (__bf16)f.x; o[1] = (__bf16)f.y; o[2] = (__bf16)f.z; o[3] = (__bf16)f.w;
  ((v4bf*)xb)[i] = o;
}

// ---------------------------------------------------------------------------
// Pre-pass 0b: wq/wk/wv [H][E][D] fp32 -> transposed [H][D][E] bf16
// (makes attention-projection B-fragments K-contiguous)
// ---------------------------------------------------------------------------
__global__ __launch_bounds__(256) void wt_qkv_kernel(
    const float* __restrict__ wq, const float* __restrict__ wk,
    const float* __restrict__ wv, __bf16* __restrict__ tq,
    __bf16* __restrict__ tk, __bf16* __restrict__ tv) {
  int i = blockIdx.x * 256 + threadIdx.x;   // over H*E*D = 147456
  int d = i % HD;
  int e = (i / HD) % EMB;
  int h = i / (HD * EMB);
  size_t dst = ((size_t)h * HD + d) * EMB + e;
  tq[dst] = (__bf16)wq[i];
  tk[dst] = (__bf16)wk[i];
  tv[dst] = (__bf16)wv[i];
}

// Pre-pass 0c: w_proj [K=384][N=384] fp32 -> transposed [N][K] bf16
__global__ __launch_bounds__(256) void wt_proj_kernel(const float* __restrict__ wp,
                                                      __bf16* __restrict__ wpT) {
  int i = blockIdx.x * 256 + threadIdx.x;   // over EMB*EMB = 147456
  int n = i % EMB;
  int k = i / EMB;
  wpT[(size_t)n * EMB + k] = (__bf16)wp[i];
}

// ---------------------------------------------------------------------------
// Kernel 1: fused QKV projection.  One wave per (b,h,t16,n16) tile position;
// Q,K,V tiles share the A fragment.  Per K-step: 2 b128 (A) + 6 b128 (B) + 3
// WMMA.  Q,K row-major [B,H,T,D] bf16; V transposed [B,H,D,T] bf16.
// ---------------------------------------------------------------------------
__global__ __launch_bounds__(128) void qkv_kernel(
    const __bf16* __restrict__ xb, const __bf16* __restrict__ tq,
    const __bf16* __restrict__ tk, const __bf16* __restrict__ tv,
    __bf16* __restrict__ Qb, __bf16* __restrict__ Kb, __bf16* __restrict__ Vt) {
  int wave = blockIdx.x * 4 + (threadIdx.x >> 5);
  int lane = threadIdx.x & 31;
  int nn = wave & 3;                 // D/16 = 4
  int tt = (wave >> 2) & 15;         // T/16 = 16
  int h  = (wave >> 6) % NH;
  int b  = wave / (4 * 16 * NH);

  const __bf16* xp = xb + ((size_t)b * TSEQ + tt * 16) * EMB;
  size_t wofs = ((size_t)h * HD + nn * 16) * EMB;   // transposed weights [h][d][e]
  const __bf16* qp = tq + wofs;
  const __bf16* kp = tk + wofs;
  const __bf16* vp = tv + wofs;

  v8f cq = {}, ck = {}, cv = {};
  for (int k0 = 0; k0 < EMB; k0 += 32) {
    v16bf a  = load_a_bf16(xp + k0, EMB, lane);
    v16bf bq = load_b_contig_bf16(qp + k0, EMB, lane);
    v16bf bk = load_b_contig_bf16(kp + k0, EMB, lane);
    v16bf bv = load_b_contig_bf16(vp + k0, EMB, lane);
    cq = wmma_bf16(a, bq, cq);
    ck = wmma_bf16(a, bk, ck);
    cv = wmma_bf16(a, bv, cv);
  }

  size_t base = (size_t)(b * NH + h) * TSEQ * HD;
  int n = lane & 15, half = lane >> 4;
#pragma unroll
  for (int v = 0; v < 8; ++v) {
    int t = tt * 16 + half * 8 + v;
    int d = nn * 16 + n;
    Qb[base + (size_t)t * HD + d] = (__bf16)cq[v];
    Kb[base + (size_t)t * HD + d] = (__bf16)ck[v];
    Vt[base + (size_t)d * TSEQ + t] = (__bf16)cv[v];   // transposed
  }
}

// ---------------------------------------------------------------------------
// Kernel 2: causal flash attention.  One wave per (b,h,16-query tile).
// Online softmax; P round-trips through a private LDS slot to convert
// C-layout f32 -> A-layout bf16 (ds_store/ds_load_b128).
// ---------------------------------------------------------------------------
__global__ __launch_bounds__(128) void attn_kernel(
    const __bf16* __restrict__ Qb, const __bf16* __restrict__ Kb,
    const __bf16* __restrict__ Vt, __bf16* __restrict__ att) {
  __shared__ alignas(16) __bf16 lds[4][16 * 32];
  int wslot = threadIdx.x >> 5;
  int lane  = threadIdx.x & 31;
  int wave  = blockIdx.x * 4 + wslot;
  int qt = wave & 15;                // T/16 = 16
  int h  = (wave >> 4) % NH;
  int b  = wave / (16 * NH);

  size_t bh = (size_t)(b * NH + h);
  const __bf16* Q = Qb + bh * TSEQ * HD;
  const __bf16* K = Kb + bh * TSEQ * HD;
  const __bf16* V = Vt + bh * (size_t)HD * TSEQ;   // [d][t]
  int q0 = qt * 16;
  int half = lane >> 4, nl = lane & 15;

  // Preload Q A-fragments for d-chunks [0,32) and [32,64).
  v16bf qa0 = load_a_bf16(Q + (size_t)q0 * HD + 0, HD, lane);
  v16bf qa1 = load_a_bf16(Q + (size_t)q0 * HD + 32, HD, lane);

  v8f o[4] = {{0}, {0}, {0}, {0}};
  float mrow[8], lrow[8];
#pragma unroll
  for (int v = 0; v < 8; ++v) { mrow[v] = -1e30f; lrow[v] = 0.f; }

  const float scale = 0.125f;        // 1/sqrt(64)
  int nchunks = (q0 + 16 + 31) / 32; // keys 0 .. q0+15
  for (int c = 0; c < nchunks; ++c) {
    int j0 = c * 32;
    // Prefetch next chunk's K rows (one row per lane) -> global_prefetch_b8
    if (c + 1 < nchunks)
      __builtin_prefetch(K + (size_t)(j0 + 32 + lane) * HD, 0, 3);

    // ---- S = Q K^T for 32 keys: two 16-key N-tiles, K-dim = 64 (2 wmma each)
    v8f s[2] = {{0}, {0}};
#pragma unroll
    for (int t = 0; t < 2; ++t) {
      v16bf kb0 = load_b_contig_bf16(K + (size_t)(j0 + t * 16) * HD + 0, HD, lane);
      v16bf kb1 = load_b_contig_bf16(K + (size_t)(j0 + t * 16) * HD + 32, HD, lane);
      s[t] = wmma_bf16(qa0, kb0, s[t]);
      s[t] = wmma_bf16(qa1, kb1, s[t]);
    }
    // ---- scale + causal mask
#pragma unroll
    for (int t = 0; t < 2; ++t) {
      int kidx = j0 + t * 16 + nl;
#pragma unroll
      for (int v = 0; v < 8; ++v) {
        int qi = q0 + half * 8 + v;
        float val = s[t][v] * scale;
        s[t][v] = (kidx <= qi) ? val : -1e30f;
      }
    }
    // ---- online softmax update (row reductions across each 16-lane half)
#pragma unroll
    for (int v = 0; v < 8; ++v) {
      float rm = fmaxf(s[0][v], s[1][v]);
#pragma unroll
      for (int off = 1; off < 16; off <<= 1)
        rm = fmaxf(rm, __shfl_xor(rm, off, 32));
      float mn = fmaxf(mrow[v], rm);
      float corr = __expf(mrow[v] - mn);
      float e0 = __expf(s[0][v] - mn);
      float e1 = __expf(s[1][v] - mn);
      float rs = e0 + e1;
#pragma unroll
      for (int off = 1; off < 16; off <<= 1)
        rs += __shfl_xor(rs, off, 32);
      lrow[v] = lrow[v] * corr + rs;
      mrow[v] = mn;
      s[0][v] = e0;
      s[1][v] = e1;
#pragma unroll
      for (int t = 0; t < 4; ++t) o[t][v] *= corr;
    }
    // ---- C-layout P -> LDS (bf16 row-major 16x32) -> A-layout fragment
    __bf16* pl = lds[wslot];
#pragma unroll
    for (int t = 0; t < 2; ++t)
#pragma unroll
      for (int v = 0; v < 8; ++v)
        pl[(half * 8 + v) * 32 + t * 16 + nl] = (__bf16)s[t][v];
    v16bf pa = load_a_bf16(pl, 32, lane);   // same-wave LDS, DS is in-order
    // ---- O += P * V  (4 N-tiles over d; V transposed -> contiguous B loads)
#pragma unroll
    for (int t = 0; t < 4; ++t) {
      v16bf vb = load_b_contig_bf16(V + (size_t)(t * 16) * TSEQ + j0, TSEQ, lane);
      o[t] = wmma_bf16(pa, vb, o[t]);
    }
  }

  // ---- normalize by l and store concat-head layout [B, T, H*D] bf16
  size_t obase = (size_t)b * TSEQ * (NH * HD) + (size_t)h * HD;
#pragma unroll
  for (int v = 0; v < 8; ++v) {
    float inv = 1.0f / lrow[v];
    int qi = q0 + half * 8 + v;
#pragma unroll
    for (int t = 0; t < 4; ++t)
      att[obase + (size_t)qi * (NH * HD) + t * 16 + nl] = (__bf16)(o[t][v] * inv);
  }
}

// ---------------------------------------------------------------------------
// Kernel 3: output projection  out = att @ w_proj + b_proj   (fp32 out)
// One wave per 16x16 output tile: M-tiles = B*T/16 = 4096, N-tiles = 24.
// Per K-step: 2 b128 (A) + 2 b128 (B, pre-transposed bf16) + 1 WMMA.
// ---------------------------------------------------------------------------
__global__ __launch_bounds__(128) void proj_kernel(
    const __bf16* __restrict__ att, const __bf16* __restrict__ wpT,
    const float* __restrict__ bp, float* __restrict__ out) {
  int wave = blockIdx.x * 4 + (threadIdx.x >> 5);
  int lane = threadIdx.x & 31;
  int nn = wave % 24;
  int mt = wave / 24;

  const __bf16* ap = att + (size_t)mt * 16 * EMB;
  const __bf16* wpp = wpT + (size_t)nn * 16 * EMB;   // [n][k], K-contiguous

  v8f c = {};
  for (int k0 = 0; k0 < EMB; k0 += 32) {
    v16bf a = load_a_bf16(ap + k0, EMB, lane);
    v16bf b = load_b_contig_bf16(wpp + k0, EMB, lane);
    c = wmma_bf16(a, b, c);
  }
  int nl = lane & 15, half = lane >> 4;
  float bias = bp[nn * 16 + nl];
#pragma unroll
  for (int v = 0; v < 8; ++v) {
    int m = mt * 16 + half * 8 + v;
    out[(size_t)m * EMB + nn * 16 + nl] = c[v] + bias;
  }
}

// ---------------------------------------------------------------------------
extern "C" void kernel_launch(void* const* d_in, const int* in_sizes, int n_in,
                              void* d_out, int out_size, void* d_ws, size_t ws_size,
                              hipStream_t stream) {
  (void)in_sizes; (void)n_in; (void)out_size; (void)ws_size;
  const float* x  = (const float*)d_in[0];
  const float* wq = (const float*)d_in[1];
  const float* wk = (const float*)d_in[2];
  const float* wv = (const float*)d_in[3];
  const float* wp = (const float*)d_in[4];
  const float* bp = (const float*)d_in[5];
  float* out = (float*)d_out;

  // Workspace layout (all bf16):
  //   Qb | Kb | Vt | att | xb  (25,165,824 elems each)
  //   tq | tk | tv | wpT       (147,456 elems each)           total ~253 MB
  const size_t nqkv = (size_t)BBATCH * NH * TSEQ * HD;
  const size_t nw   = (size_t)NH * EMB * HD;                   // 147456
  __bf16* Qb  = (__bf16*)d_ws;
  __bf16* Kb  = Qb + nqkv;
  __bf16* Vt  = Kb + nqkv;
  __bf16* att = Vt + nqkv;
  __bf16* xb  = att + nqkv;
  __bf16* tq  = xb + nqkv;
  __bf16* tk  = tq + nw;
  __bf16* tv  = tk + nw;
  __bf16* wpT = tv + nw;

  // Pre-passes: convert / transpose into WMMA-friendly bf16 layouts.
  cvt_x_kernel<<<(BBATCH * TSEQ * EMB / 4) / 256, 256, 0, stream>>>(x, xb);
  wt_qkv_kernel<<<(NH * EMB * HD) / 256, 256, 0, stream>>>(wq, wk, wv, tq, tk, tv);
  wt_proj_kernel<<<(EMB * EMB) / 256, 256, 0, stream>>>(wp, wpT);

  // Kernel 1: B*H*(T/16)*(D/16) = 98304 waves, 4 waves/block
  qkv_kernel<<<98304 / 4, 128, 0, stream>>>(xb, tq, tk, tv, Qb, Kb, Vt);
  // Kernel 2: B*H*(T/16) = 24576 waves
  attn_kernel<<<24576 / 4, 128, 0, stream>>>(Qb, Kb, Vt, att);
  // Kernel 3: (B*T/16)*(EMB/16) = 98304 waves
  proj_kernel<<<98304 / 4, 128, 0, stream>>>(att, wpT, bp, out);
}
